// MultiheadAttention_41661182771260
// MI455X (gfx1250) — matrix operand order repaired and checked by reading, not compiled, not run on hardware
//
#include <hip/hip_runtime.h>
#include <math.h>

typedef _Float16 f16;
typedef __attribute__((ext_vector_type(16))) _Float16 v16h;
typedef __attribute__((ext_vector_type(8)))  _Float16 v8h;
typedef __attribute__((ext_vector_type(8)))  float    v8f;

namespace {
constexpr int kB  = 4;
constexpr int kS  = 2048;
constexpr int kD  = 512;
constexpr int kH  = 8;
constexpr int kHD = 64;
constexpr int kM  = kB * kS;   // 8192 rows for the projection GEMMs
constexpr int LSTR  = 48;      // 32-col GEMM LDS tile stride in halves (96B)
constexpr int LSTRK = 72;      // 64-col K/V/P LDS tile stride in halves (144B)
}

__device__ inline v16h cat16(v8h lo, v8h hi) {
  v16h r;
#pragma unroll
  for (int i = 0; i < 8; ++i) { r[i] = lo[i]; r[i + 8] = hi[i]; }
  return r;
}

// A fragment (16x32 f16, M x K) from a row-major f16 tile.
// lane<16: row m=lane, K = 0..7 / 16..23; lane>=16: same row, K = 8..15 / 24..31.
__device__ inline v16h load_a_frag(const f16* base, int lane, int stride) {
  const f16* p = base + (size_t)(lane & 15) * stride + (lane >> 4) * 8;
  return cat16(*(const v8h*)p, *(const v8h*)(p + 16));
}

// B fragment (32x16, K x N); tile rows indexed by N, cols by K
// (B[k][n] = rowmajor[n][k]). lanes 0-15: K=0..7 / 16..23; lanes 16-31: K=8..15 / 24..31.
__device__ inline v16h load_b_frag(const f16* base, int lane, int stride) {
  const f16* p = base + (size_t)(lane & 15) * stride + (lane >> 4) * 8;
  return cat16(*(const v8h*)p, *(const v8h*)(p + 16));
}

__device__ inline v8f wmma16(v16h a, v16h b, v8f c) {
  return __builtin_amdgcn_wmma_f32_16x16x32_f16(false, a, false, b, (short)0, c,
                                                false, false);
}

// Async 16B global -> LDS copy (ASYNCcnt-tracked, bypasses VGPRs).
__device__ inline void async_copy_b128(f16* lds, const f16* g) {
  const unsigned lo = (unsigned)(size_t)lds;        // LDS offset = addr[31:0]
  const unsigned long long ga = (unsigned long long)(size_t)g;
  asm volatile("global_load_async_to_lds_b128 %0, %1, off"
               :: "v"(lo), "v"(ga) : "memory");
}
__device__ inline void wait_async0() {
  asm volatile("s_wait_asynccnt 0x0" ::: "memory");
}

// ---------------------------------------------------------------------------
// Kernel 1: fused Q/K/V projections. y = X @ W^T + b, fp32 in, f16 out.
// blockIdx.z selects {Q,K,V}. Q,K stored [B,H,S,HD]; V stored transposed
// [B,H,HD,S] so attention PV B-fragments are contiguous.
// ---------------------------------------------------------------------------
__global__ __launch_bounds__(128)
void qkv_proj_kernel(const float* __restrict__ Xq, const float* __restrict__ Xk,
                     const float* __restrict__ Xv,
                     const float* __restrict__ Wq, const float* __restrict__ Wk,
                     const float* __restrict__ Wv,
                     const float* __restrict__ bq, const float* __restrict__ bk,
                     const float* __restrict__ bv,
                     f16* __restrict__ Qo, f16* __restrict__ Ko,
                     f16* __restrict__ Vt) {
  __shared__ f16 Xs[64 * LSTR];
  __shared__ f16 Ws[64 * LSTR];

  const int z = blockIdx.z;
  const float* X    = (z == 0) ? Xq : (z == 1) ? Xk : Xv;
  const float* W    = (z == 0) ? Wq : (z == 1) ? Wk : Wv;
  const float* bias = (z == 0) ? bq : (z == 1) ? bk : bv;

  const int tid  = threadIdx.x;
  const int lane = tid & 31;
  const int wid  = tid >> 5;
  const int waveM = (wid >> 1) * 32;
  const int waveN = (wid & 1) * 32;
  const int m0 = blockIdx.x * 64;
  const int n0 = blockIdx.y * 64;

  v8f acc[2][2] = {};

  for (int k0 = 0; k0 < kD; k0 += 32) {
    __syncthreads();
    // Stage X and W tiles (64x32 each) fp32 -> f16 into LDS.
    // Exact trip counts (512 chunks / 128 threads = 4) -> no EXEC predication.
#pragma unroll
    for (int it = 0; it < 4; ++it) {
      const int i = tid + it * 128;
      const int r = i >> 3, c = (i & 7) * 4;
      const float4 vx = *(const float4*)(X + (size_t)(m0 + r) * kD + k0 + c);
      f16* px = Xs + r * LSTR + c;
      px[0] = (f16)vx.x; px[1] = (f16)vx.y; px[2] = (f16)vx.z; px[3] = (f16)vx.w;
      const float4 vw = *(const float4*)(W + (size_t)(n0 + r) * kD + k0 + c);
      f16* pw = Ws + r * LSTR + c;
      pw[0] = (f16)vw.x; pw[1] = (f16)vw.y; pw[2] = (f16)vw.z; pw[3] = (f16)vw.w;
    }
    __syncthreads();

    v16h aF[2], bF[2];
#pragma unroll
    for (int mi = 0; mi < 2; ++mi)
      aF[mi] = load_a_frag(Xs + (waveM + mi * 16) * LSTR, lane, LSTR);
#pragma unroll
    for (int ni = 0; ni < 2; ++ni)
      bF[ni] = load_b_frag(Ws + (waveN + ni * 16) * LSTR, lane, LSTR);
#pragma unroll
    for (int mi = 0; mi < 2; ++mi)
#pragma unroll
      for (int ni = 0; ni < 2; ++ni)
        acc[mi][ni] = wmma16(aF[mi], bF[ni], acc[mi][ni]);
  }

  // Epilogue: bias + f16 store. C layout: lane holds rows (lane>>4)*8+r, col lane&15.
#pragma unroll
  for (int mi = 0; mi < 2; ++mi) {
#pragma unroll
    for (int ni = 0; ni < 2; ++ni) {
      const int n  = n0 + waveN + ni * 16 + (lane & 15);
      const int mb = m0 + waveM + mi * 16 + (lane >> 4) * 8;
      const float bn = bias[n];
      const int h = n >> 6, d = n & 63;
      if (z < 2) {
        f16* dst = (z == 0) ? Qo : Ko;
#pragma unroll
        for (int r = 0; r < 8; ++r) {
          const int m = mb + r;
          const int bb = m >> 11, ss = m & (kS - 1);
          dst[(((size_t)bb * kH + h) * kS + ss) * kHD + d] =
              (f16)(acc[mi][ni][r] + bn);
        }
      } else {
        const int bb = mb >> 11, ss = mb & (kS - 1);  // 8 consecutive s
        v8h pv;
#pragma unroll
        for (int r = 0; r < 8; ++r) pv[r] = (f16)(acc[mi][ni][r] + bn);
        *(v8h*)(Vt + (((size_t)bb * kH + h) * kHD + d) * kS + ss) = pv;
      }
    }
  }
}

// ---------------------------------------------------------------------------
// Kernel 2: flash attention. Block = 4 waves; wave owns 16 query rows.
// K/V tiles (shared by all 4 waves) staged global->LDS with async copies.
// Q,K: [B,H,S,HD] f16. Vt: [B,H,HD,S] f16. Ctx out: [B,S,D] f16.
// ---------------------------------------------------------------------------
__global__ __launch_bounds__(128)
void attn_kernel(const f16* __restrict__ Q, const f16* __restrict__ K,
                 const f16* __restrict__ Vt, f16* __restrict__ Ctx) {
  __shared__ f16 Ks[64 * LSTRK];     // rows = key index within tile, cols = d
  __shared__ f16 Vs[64 * LSTRK];     // rows = d, cols = s within tile
  __shared__ f16 Ps[4][16 * LSTRK];  // per-wave P transpose slab

  const int tid  = threadIdx.x;
  const int lane = tid & 31;
  const int w    = tid >> 5;
  const int bh   = blockIdx.y;                  // b*H + h
  const int q0   = blockIdx.x * 64 + w * 16;    // this wave's query base

  const f16* Qh = Q  + (size_t)bh * kS * kHD;
  const f16* Kh = K  + (size_t)bh * kS * kHD;
  const f16* Vh = Vt + (size_t)bh * kHD * kS;

  // Q fragments for this wave's 16 rows (HD=64 -> two K=32 chunks), kept live.
  v16h qf[2];
#pragma unroll
  for (int kc = 0; kc < 2; ++kc)
    qf[kc] = load_a_frag(Qh + (size_t)q0 * kHD + kc * 32, lane, kHD);

  float rowMax[8], rowSum[8];
#pragma unroll
  for (int r = 0; r < 8; ++r) { rowMax[r] = -3.0e30f; rowSum[r] = 0.0f; }
  v8f acc[4] = {};
  const float scale = 0.125f;  // 1/sqrt(HD)

  for (int kt = 0; kt < kS; kt += 64) {
    __syncthreads();  // previous tile fully consumed by all waves
    // ---- async-stage K (64x64) and V (64x64) tiles, 16B per op ----
#pragma unroll
    for (int it = 0; it < 4; ++it) {
      const int i = tid + it * 128;          // 0..511
      const int r = i >> 3, c = (i & 7) * 8; // row, col(halves)
      async_copy_b128(Ks + r * LSTRK + c, Kh + (size_t)(kt + r) * kHD + c);
      async_copy_b128(Vs + r * LSTRK + c, Vh + (size_t)r * kS + kt + c);
    }
    wait_async0();    // own async group done
    __syncthreads();  // everyone's async group done

    // ---- S = Q K^T * scale (16x64 per wave) ----
    v8f sf[4];
#pragma unroll
    for (int ni = 0; ni < 4; ++ni) {
      v8f c = {};
#pragma unroll
      for (int kc = 0; kc < 2; ++kc)
        c = wmma16(qf[kc],
                   load_b_frag(Ks + (ni * 16) * LSTRK + kc * 32, lane, LSTRK),
                   c);
      sf[ni] = c;
    }

    // ---- online softmax (row lives in a 16-lane half, per C-layout) ----
    float corr[8];
#pragma unroll
    for (int r = 0; r < 8; ++r) {
      float mx = sf[0][r] * scale;
#pragma unroll
      for (int ni = 1; ni < 4; ++ni) mx = fmaxf(mx, sf[ni][r] * scale);
#pragma unroll
      for (int off = 1; off < 16; off <<= 1)
        mx = fmaxf(mx, __shfl_xor(mx, off, 32));
      const float nm = fmaxf(rowMax[r], mx);
      corr[r] = __expf(rowMax[r] - nm);
      rowMax[r] = nm;
    }
#pragma unroll
    for (int r = 0; r < 8; ++r) {
      float ps = 0.0f;
#pragma unroll
      for (int ni = 0; ni < 4; ++ni) {
        const float p = __expf(sf[ni][r] * scale - rowMax[r]);
        sf[ni][r] = p;
        ps += p;
      }
#pragma unroll
      for (int off = 1; off < 16; off <<= 1) ps += __shfl_xor(ps, off, 32);
      rowSum[r] = rowSum[r] * corr[r] + ps;
#pragma unroll
      for (int di = 0; di < 4; ++di) acc[di][r] *= corr[r];
    }

    // ---- C-layout -> A-layout transpose of P through per-wave LDS ----
    f16* pb = &Ps[w][0];
#pragma unroll
    for (int ni = 0; ni < 4; ++ni) {
      const int n = lane & 15, mb = (lane >> 4) * 8;
#pragma unroll
      for (int r = 0; r < 8; ++r)
        pb[(mb + r) * LSTRK + ni * 16 + n] = (f16)sf[ni][r];
    }
    asm volatile("s_wait_dscnt 0x0" ::: "memory");  // wave-local DS is in-order

    v16h pf[2];
#pragma unroll
    for (int kc = 0; kc < 2; ++kc)
      pf[kc] = load_a_frag(pb + kc * 32, lane, LSTRK);

    // ---- ctx += P @ V ----
#pragma unroll
    for (int di = 0; di < 4; ++di)
#pragma unroll
      for (int kc = 0; kc < 2; ++kc)
        acc[di] = wmma16(pf[kc],
                         load_b_frag(Vs + (di * 16) * LSTRK + kc * 32, lane,
                                     LSTRK),
                         acc[di]);
  }

  // ---- normalize and merge heads: Ctx[b][s][h*HD + d] ----
  const int bb = bh >> 3, h = bh & 7;
#pragma unroll
  for (int di = 0; di < 4; ++di) {
    const int d  = h * kHD + di * 16 + (lane & 15);
    const int mb = q0 + (lane >> 4) * 8;
#pragma unroll
    for (int r = 0; r < 8; ++r) {
      Ctx[((size_t)bb * kS + (mb + r)) * kD + d] =
          (f16)(acc[di][r] / rowSum[r]);
    }
  }
}

// ---------------------------------------------------------------------------
// Kernel 3: output projection. y = Ctx(f16) @ Wo^T + bo, fp32 out.
// Ctx tile is already f16 -> staged with async global->LDS copies.
// ---------------------------------------------------------------------------
__global__ __launch_bounds__(128)
void out_proj_kernel(const f16* __restrict__ X, const float* __restrict__ W,
                     const float* __restrict__ bias, float* __restrict__ Y) {
  __shared__ f16 Xs[64 * LSTR];
  __shared__ f16 Ws[64 * LSTR];

  const int tid  = threadIdx.x;
  const int lane = tid & 31;
  const int wid  = tid >> 5;
  const int waveM = (wid >> 1) * 32;
  const int waveN = (wid & 1) * 32;
  const int m0 = blockIdx.x * 64;
  const int n0 = blockIdx.y * 64;

  v8f acc[2][2] = {};

  for (int k0 = 0; k0 < kD; k0 += 32) {
    __syncthreads();
    // X tile: 256 16B chunks async to LDS (2 per thread).
#pragma unroll
    for (int it = 0; it < 2; ++it) {
      const int i = tid + it * 128;
      const int r = i >> 2, c = (i & 3) * 8;
      async_copy_b128(Xs + r * LSTR + c, X + (size_t)(m0 + r) * kD + k0 + c);
    }
    // W tile fp32 -> f16 via VGPRs.
#pragma unroll
    for (int it = 0; it < 4; ++it) {
      const int i = tid + it * 128;
      const int r = i >> 3, c = (i & 7) * 4;
      const float4 v = *(const float4*)(W + (size_t)(n0 + r) * kD + k0 + c);
      f16* p = Ws + r * LSTR + c;
      p[0] = (f16)v.x; p[1] = (f16)v.y; p[2] = (f16)v.z; p[3] = (f16)v.w;
    }
    wait_async0();
    __syncthreads();

    v16h aF[2], bF[2];
#pragma unroll
    for (int mi = 0; mi < 2; ++mi)
      aF[mi] = load_a_frag(Xs + (waveM + mi * 16) * LSTR, lane, LSTR);
#pragma unroll
    for (int ni = 0; ni < 2; ++ni)
      bF[ni] = load_b_frag(Ws + (waveN + ni * 16) * LSTR, lane, LSTR);
#pragma unroll
    for (int mi = 0; mi < 2; ++mi)
#pragma unroll
      for (int ni = 0; ni < 2; ++ni)
        acc[mi][ni] = wmma16(aF[mi], bF[ni], acc[mi][ni]);
  }

#pragma unroll
  for (int mi = 0; mi < 2; ++mi) {
#pragma unroll
    for (int ni = 0; ni < 2; ++ni) {
      const int n  = n0 + waveN + ni * 16 + (lane & 15);
      const int mb = m0 + waveM + mi * 16 + (lane >> 4) * 8;
      const float bn = bias[n];
#pragma unroll
      for (int r = 0; r < 8; ++r)
        Y[(size_t)(mb + r) * kD + n] = acc[mi][ni][r] + bn;
    }
  }
}

// ---------------------------------------------------------------------------
// Launch. Workspace layout (f16): Q | K | Vt | Ctx, 8 MB each (32 MB total).
// ---------------------------------------------------------------------------
extern "C" void kernel_launch(void* const* d_in, const int* in_sizes, int n_in,
                              void* d_out, int out_size, void* d_ws,
                              size_t ws_size, hipStream_t stream) {
  (void)in_sizes; (void)n_in; (void)out_size; (void)ws_size;
  const float* q  = (const float*)d_in[0];
  const float* k  = (const float*)d_in[1];
  const float* v  = (const float*)d_in[2];
  const float* Wq = (const float*)d_in[3];
  const float* bq = (const float*)d_in[4];
  const float* Wk = (const float*)d_in[5];
  const float* bk = (const float*)d_in[6];
  const float* Wv = (const float*)d_in[7];
  const float* bv = (const float*)d_in[8];
  const float* Wo = (const float*)d_in[9];
  const float* bo = (const float*)d_in[10];
  float* out = (float*)d_out;

  const size_t elems = (size_t)kB * kS * kD;  // 4,194,304
  f16* Qh  = (f16*)d_ws;
  f16* Kh  = Qh + elems;
  f16* Vt  = Kh + elems;
  f16* Ctx = Vt + elems;

  qkv_proj_kernel<<<dim3(kM / 64, kD / 64, 3), 128, 0, stream>>>(
      q, k, v, Wq, Wk, Wv, bq, bk, bv, Qh, Kh, Vt);
  attn_kernel<<<dim3(kS / 64, kB * kH), 128, 0, stream>>>(Qh, Kh, Vt, Ctx);
  out_proj_kernel<<<dim3(kM / 64, kD / 64), 128, 0, stream>>>(Ctx, Wo, bo, out);
}
